// CrossViewAttention_90426241450587
// MI455X (gfx1250) — matrix operand order repaired and checked by reading, not compiled, not run on hardware
//
#include <hip/hip_runtime.h>
#include <hip/hip_bf16.h>

#define NVIEW 8
#define NHEAD 8
#define DIM   512
#define HDIM  64
#define SEQ   2048
#define BATCH 4

typedef _Float16 half_t;
typedef __attribute__((ext_vector_type(8)))  _Float16 v8h;
typedef __attribute__((ext_vector_type(16))) _Float16 v16h;
typedef __attribute__((ext_vector_type(8)))  float    v8f;

__device__ __forceinline__ v16h cat8(v8h lo, v8h hi) {
    return __builtin_shufflevector(lo, hi, 0,1,2,3,4,5,6,7,8,9,10,11,12,13,14,15);
}
__device__ __forceinline__ v8f wmma_f16(v16h a, v16h b, v8f c) {
    // D = A(16x32 f16) * B(32x16 f16) + C(16x16 f32)
    return __builtin_amdgcn_wmma_f32_16x16x32_f16(
        /*neg_a=*/false, a, /*neg_b=*/false, b,
        /*c_mod=*/(short)0, c, /*reuse_a=*/false, /*reuse_b=*/false);
}
// A fragment: lane holds row (ln&15); K interleave: lo-lanes K {0..7,16..23}, hi-lanes {8..15,24..31}
__device__ __forceinline__ v16h load_a_frag(const half_t* rowptr_kk_koff) {
    v8h a0 = *(const v8h*)(rowptr_kk_koff);
    v8h a1 = *(const v8h*)(rowptr_kk_koff + 16);
    return cat8(a0, a1);
}
// B fragment (from W^T row-major, i.e. [col][k]): lane holds col (ln&15); K interleave: lo-lanes K {0..7,16..23}, hi-lanes {8..15,24..31}
__device__ __forceinline__ v16h load_b_frag(const half_t* colptr_kk_kboff) {
    v8h b0 = *(const v8h*)(colptr_kk_kboff);
    v8h b1 = *(const v8h*)(colptr_kk_kboff + 16);
    return cat8(b0, b1);
}

// ---------------- Kernel 1: column-sum of x over S (for kmean) + fp32->f16 convert ----------------
__global__ __launch_bounds__(256)
void k1_xsum_cvt(const float* __restrict__ x, half_t* __restrict__ xh, float* __restrict__ xsum) {
    int blk = blockIdx.x;          // 32 bv * 2 dchunk * 16 schunk = 1024
    int sc  = blk & 15;
    int dc  = (blk >> 4) & 1;
    int bv  = blk >> 5;
    int d   = dc * 256 + threadIdx.x;
    size_t base = ((size_t)bv * SEQ + (size_t)sc * 128) * DIM + d;
    float acc = 0.f;
    for (int s = 0; s < 128; ++s) {
        // x is touched exactly once in the whole pipeline -> non-temporal
        float v = __builtin_nontemporal_load(&x[base + (size_t)s * DIM]);
        acc += v;
        xh[base + (size_t)s * DIM] = (half_t)v;
    }
    atomicAdd(&xsum[bv * DIM + d], acc);
}

// ---------------- Kernel P0: Wv, Wo -> f16 transposed (W^T[n][k] = W[k][n]) ----------------
__global__ __launch_bounds__(512)
void p0_wt(const float* __restrict__ Wv, const float* __restrict__ Wo,
           half_t* __restrict__ WvT, half_t* __restrict__ WoT) {
    int blk = blockIdx.x;                       // 1024: [mat*512 + n]
    const float* W  = (blk < 512) ? Wv : Wo;
    half_t*      WT = (blk < 512) ? WvT : WoT;
    int n = blk & 511;
    int k = threadIdx.x;
    WT[(size_t)n * DIM + k] = (half_t)W[(size_t)k * DIM + n];
}

// ---------------- Kernel 2a: kmean[32][512] = (xsum/2048) @ Wk + bk ----------------
__global__ __launch_bounds__(512)
void k2a_kmean(const float* __restrict__ xsum, const float* __restrict__ Wk,
               const float* __restrict__ bk, float* __restrict__ kmean) {
    __shared__ float xm[DIM];
    int bv = blockIdx.x, d = threadIdx.x;
    xm[d] = xsum[bv * DIM + d] * (1.0f / (float)SEQ);
    __syncthreads();
    float acc = bk[d];
    for (int k = 0; k < DIM; ++k) acc += xm[k] * Wk[k * DIM + d];
    kmean[bv * DIM + d] = acc;
}

// ---------------- Kernel 2b: U^T[b][c=j*8+h][k] = sum_hd Wq[k][h*64+hd]*kmean[b,j,h,hd]; bqdot ----------------
__global__ __launch_bounds__(512)
void k2b_ut(const float* __restrict__ kmean, const float* __restrict__ Wq,
            const float* __restrict__ bq, half_t* __restrict__ UT, float* __restrict__ bqdot) {
    __shared__ float km[HDIM];
    int blk = blockIdx.x;                      // 256: b*64 + c
    int b = blk >> 6, c = blk & 63;
    int j = c >> 3, h = c & 7;
    int k = threadIdx.x;
    if (k < HDIM) km[k] = kmean[(b * NVIEW + j) * DIM + h * HDIM + k];
    __syncthreads();
    float acc = 0.f;
    #pragma unroll 8
    for (int hd = 0; hd < HDIM; ++hd) acc += Wq[(size_t)k * DIM + h * HDIM + hd] * km[hd];
    UT[((size_t)b * 64 + c) * DIM + k] = (half_t)acc;
    if (k == 0) {
        float a2 = 0.f;
        for (int hd = 0; hd < HDIM; ++hd) a2 += bq[h * HDIM + hd] * km[hd];
        bqdot[b * 64 + c] = a2;
    }
}

// ---------------- Kernel 3: fused scores -> softmax -> V -> combine -> Y, per (b, 16-pos s-tile) ----------------
#define XV_STRIDE 520   // 512 + 8 f16 pad (16B) -> 4-bank rotation per row, conflict-free b128
#define W_STRIDE  65    // scores/weights row stride (f32)
#define LDS_BYTES (128 * XV_STRIDE * 2 + 128 * W_STRIDE * 4)   // 133120 + 33280 = 166400

__global__ __launch_bounds__(512)
void k3_fused(const half_t* __restrict__ xh,    // [32][2048][512] f16
              const half_t* __restrict__ UT,    // [4][64][512]    f16
              const float*  __restrict__ bqdot, // [4][64]
              const half_t* __restrict__ WvT,   // [512][512]      f16 (W^T)
              const float*  __restrict__ bvv,
              const half_t* __restrict__ WoT,   // [512][512]      f16 (W^T)
              const float*  __restrict__ bo,
              float* __restrict__ y) {          // [32][2048][512] f32
    extern __shared__ char smem[];
    half_t* ldsXV = (half_t*)smem;                               // [128][XV_STRIDE] : X tile, later V tile
    float*  ldsW  = (float*)(smem + 128 * XV_STRIDE * 2);        // [128][W_STRIDE]  : scores -> softmax w

    int wg = blockIdx.x;            // 512 = 4 b * 128 s-tiles
    int b  = wg >> 7;
    int s0 = (wg & 127) << 4;

    int tid  = threadIdx.x;
    int wave = tid >> 5;            // 0..15
    int ln   = tid & 31;
    int lrow = ln & 15;
    int koff = (ln >> 4) << 3;      // A-frag K interleave base (0 or 8)
    int kboff = (ln >> 4) << 3;     // B-frag K interleave base (0 or 8)

    // ---- Stage 0: warm caches with the f16 weight matrices (1 MB, reused by all 512 WGs) ----
    {
        #pragma unroll
        for (int i = 0; i < 8; ++i) {
            __builtin_prefetch(WvT + (size_t)(tid + i * 512) * 64, 0, 1);  // 128B per line
            __builtin_prefetch(WoT + (size_t)(tid + i * 512) * 64, 0, 1);
        }
    }

    // ---- Stage 1: X tile (8 views x 16 positions x 512) -> LDS via async copy (ASYNCcnt path) ----
    {
        int r = tid >> 2, q = tid & 3;          // row 0..127, 128-f16 quarter
        int view = r >> 4, t = r & 15;
        const half_t* src = xh + (((size_t)(b * NVIEW + view)) * SEQ + (s0 + t)) * DIM + q * 128;
        unsigned lds_off = (unsigned)(uintptr_t)(ldsXV + r * XV_STRIDE + q * 128);
        // INST_OFFSET is added to BOTH the global and the LDS address (ISA 08 §4.4),
        // so one base pair + 16 immediate offsets moves this thread's 256B.
        #pragma unroll
        for (int i = 0; i < 16; ++i) {
            asm volatile("global_load_async_to_lds_b128 %0, %1, off offset:%2"
                         :: "v"(lds_off), "v"(src), "i"(i * 16) : "memory");
        }
        asm volatile("s_wait_asynccnt 0x0" ::: "memory");
    }
    __syncthreads();

    // ---- Stage 2: raw scores [128][64] = X @ U_b  (wave: mtile=wave&7, 2 ntiles) ----
    {
        int mtile = wave & 7;
        const half_t* Ub = UT + (size_t)b * 64 * DIM;
        const half_t* arow = ldsXV + (mtile * 16 + lrow) * XV_STRIDE + koff;
        for (int nt = 0; nt < 2; ++nt) {
            int ntile = (wave >> 3) * 2 + nt;
            const half_t* bcol = Ub + (size_t)(ntile * 16 + lrow) * DIM + kboff;
            v8f c = {};
            #pragma unroll
            for (int kk = 0; kk < DIM; kk += 32)
                c = wmma_f16(load_a_frag(arow + kk), load_b_frag(bcol + kk), c);
            #pragma unroll
            for (int g = 0; g < 8; ++g)
                ldsW[(mtile * 16 + g + koff) * W_STRIDE + ntile * 16 + lrow] = c[g];
        }
    }
    __syncthreads();

    // ---- Stage 3: softmax over key-view axis j for each (row r, head h) ----
    {
        const float scale = 0.125f;   // HD^-0.5
        #pragma unroll
        for (int rep = 0; rep < 2; ++rep) {
            int g = tid + rep * 512;  // 1024 groups
            int r = g >> 3, h = g & 7;
            float vals[8], mx = -1e30f;
            #pragma unroll
            for (int j = 0; j < 8; ++j) {
                float sc = (ldsW[r * W_STRIDE + j * 8 + h] + bqdot[b * 64 + j * 8 + h]) * scale;
                vals[j] = sc; mx = fmaxf(mx, sc);
            }
            float sum = 0.f;
            #pragma unroll
            for (int j = 0; j < 8; ++j) { vals[j] = __expf(vals[j] - mx); sum += vals[j]; }
            float inv = 1.0f / sum;
            #pragma unroll
            for (int j = 0; j < 8; ++j) ldsW[r * W_STRIDE + j * 8 + h] = vals[j] * inv;
        }
    }
    // (no barrier needed yet: stage 4 touches only the X region; ldsW barrier is inside stage 4)

    // ---- Stage 4: V = X @ Wv + bv, accumulate in regs, then overwrite X region with f16 V ----
    {
        int mtile = wave & 7;         // view j rows
        int nh = wave >> 3;           // column half
        v8f acc[16] = {};
        const half_t* arow = ldsXV + (mtile * 16 + lrow) * XV_STRIDE + koff;
        for (int kk = 0; kk < DIM; kk += 32) {
            v16h A = load_a_frag(arow + kk);
            #pragma unroll
            for (int nt = 0; nt < 16; ++nt) {
                int ncol = nh * 256 + nt * 16 + lrow;
                v16h B = load_b_frag(WvT + (size_t)ncol * DIM + kboff + kk);
                acc[nt] = wmma_f16(A, B, acc[nt]);
            }
        }
        __syncthreads();   // all waves done reading X; stage-3 ldsW writes also ordered
        #pragma unroll
        for (int nt = 0; nt < 16; ++nt) {
            int ncol = nh * 256 + nt * 16 + lrow;
            float bb = bvv[ncol];
            #pragma unroll
            for (int g = 0; g < 8; ++g)
                ldsXV[(mtile * 16 + g + koff) * XV_STRIDE + ncol] = (half_t)(acc[nt][g] + bb);
        }
    }
    __syncthreads();

    // ---- Stage 5: Y = out @ Wo + bo, with out A-fragments built on the fly from w * V ----
    {
        int iview = wave & 7;         // query view = output rows
        int nh = wave >> 3;
        int t = lrow;                 // s position within tile (A-row local index)
        v8f acc[16] = {};
        const float* wrow = ldsW + (iview * 16 + t) * W_STRIDE;
        for (int kk = 0; kk < DIM; kk += 32) {
            int h = kk >> 6;          // head is constant within a 32-wide column chunk
            float w8[8];
            #pragma unroll
            for (int j = 0; j < 8; ++j) w8[j] = wrow[j * 8 + h];
            float o[16];
            #pragma unroll
            for (int p = 0; p < 16; ++p) o[p] = 0.f;
            #pragma unroll
            for (int j = 0; j < 8; ++j) {
                const half_t* vp = ldsXV + (j * 16 + t) * XV_STRIDE + kk + koff;
                v8h v0 = *(const v8h*)(vp);
                v8h v1 = *(const v8h*)(vp + 16);
                float wj = w8[j];
                #pragma unroll
                for (int p = 0; p < 8; ++p) {
                    o[p]     += wj * (float)v0[p];
                    o[p + 8] += wj * (float)v1[p];
                }
            }
            v16h A;
            #pragma unroll
            for (int p = 0; p < 16; ++p) A[p] = (half_t)o[p];
            #pragma unroll
            for (int nt = 0; nt < 16; ++nt) {
                int ncol = nh * 256 + nt * 16 + lrow;
                v16h B = load_b_frag(WoT + (size_t)ncol * DIM + kk + kboff);
                acc[nt] = wmma_f16(A, B, acc[nt]);
            }
        }
        size_t rowbase = ((size_t)(b * NVIEW + iview)) * SEQ + s0;
        #pragma unroll
        for (int nt = 0; nt < 16; ++nt) {
            int d = nh * 256 + nt * 16 + lrow;
            float bod = bo[d];
            #pragma unroll
            for (int g = 0; g < 8; ++g)   // y is write-once streaming output -> non-temporal
                __builtin_nontemporal_store(acc[nt][g] + bod, &y[(rowbase + g + koff) * DIM + d]);
        }
    }
}

extern "C" void kernel_launch(void* const* d_in, const int* in_sizes, int n_in,
                              void* d_out, int out_size, void* d_ws, size_t ws_size,
                              hipStream_t stream) {
    (void)in_sizes; (void)n_in; (void)out_size; (void)ws_size;
    const float* x   = (const float*)d_in[0];
    const float* Wq  = (const float*)d_in[1];
    const float* bq  = (const float*)d_in[2];
    const float* Wk  = (const float*)d_in[3];
    const float* bk  = (const float*)d_in[4];
    const float* Wv  = (const float*)d_in[5];
    const float* bv  = (const float*)d_in[6];
    const float* Wo  = (const float*)d_in[7];
    const float* bo  = (const float*)d_in[8];
    float* y = (float*)d_out;

    char* ws = (char*)d_ws;
    half_t* xh    = (half_t*)ws;  ws += (size_t)32 * SEQ * DIM * sizeof(half_t);   // 64 MB
    half_t* WvT   = (half_t*)ws;  ws += (size_t)DIM * DIM * sizeof(half_t);
    half_t* WoT   = (half_t*)ws;  ws += (size_t)DIM * DIM * sizeof(half_t);
    half_t* UT    = (half_t*)ws;  ws += (size_t)BATCH * 64 * DIM * sizeof(half_t);
    float*  xsum  = (float*)ws;   ws += (size_t)32 * DIM * sizeof(float);
    float*  kmean = (float*)ws;   ws += (size_t)32 * DIM * sizeof(float);
    float*  bqdot = (float*)ws;   ws += (size_t)BATCH * 64 * sizeof(float);

    (void)hipFuncSetAttribute((const void*)k3_fused,
                              hipFuncAttributeMaxDynamicSharedMemorySize, LDS_BYTES);

    hipMemsetAsync(xsum, 0, (size_t)32 * DIM * sizeof(float), stream);
    k1_xsum_cvt<<<1024, 256, 0, stream>>>(x, xh, xsum);
    p0_wt<<<1024, 512, 0, stream>>>(Wv, Wo, WvT, WoT);
    k2a_kmean<<<32, 512, 0, stream>>>(xsum, Wk, bk, kmean);
    k2b_ut<<<256, 512, 0, stream>>>(kmean, Wq, bq, UT, bqdot);
    k3_fused<<<512, 512, LDS_BYTES, stream>>>(xh, UT, bqdot, WvT, bv, WoT, bo, y);
}
